// MetaLearner_23347442221348
// MI455X (gfx1250) — matrix operand order repaired and checked by reading, not compiled, not run on hardware
//
#include <hip/hip_runtime.h>

typedef __attribute__((ext_vector_type(16))) _Float16 v16h;
typedef __attribute__((ext_vector_type(8)))  float    v8f;

#define Hh   20
#define FH   80
#define NT   5          // 5 tiles of 16 covering the 80 gate dims
#define WAVES 4
#define BS   128        // 4 wave32s
#define ROWS_PER_BLOCK 64
#define LN_EPS 1e-5f

// fast transcendentals: post-LN gate values are bounded O(1..5), so hardware
// v_exp_f32 / v_rcp_f32 approximations are plenty; formulas saturate correctly.
__device__ __forceinline__ float rcp_fast(float x) { return __builtin_amdgcn_rcpf(x); }
__device__ __forceinline__ float sigm(float x)     { return rcp_fast(1.0f + __expf(-x)); }
__device__ __forceinline__ float tanh_fast(float x)
{
    float e = __expf(2.0f * x);              // +inf for large x, 0 for very negative x
    return 1.0f - 2.0f * rcp_fast(e + 1.0f); // -> +1 / -1 saturation, correct limits
}

__global__ __launch_bounds__(BS)
void meta_lstm_kernel(const float* __restrict__ x,
                      const float* __restrict__ W1,  const float* __restrict__ b1,
                      const float* __restrict__ Wih, const float* __restrict__ Whh,
                      const float* __restrict__ b_ih,const float* __restrict__ b_hh,
                      const float* __restrict__ g_x, const float* __restrict__ be_x,
                      const float* __restrict__ g_h, const float* __restrict__ be_h,
                      const float* __restrict__ g_c, const float* __restrict__ be_c,
                      const float* __restrict__ Wo,  const float* __restrict__ bo,
                      float* __restrict__ out)
{
    // Weight A-fragments in exact WMMA lane layout: mats 0=Wih[0], 1=Wih[1], 2=Whh[1]
    __shared__ __align__(32) _Float16 sFrag[3 * NT * 32 * 16];   // 15 KB
    __shared__ float sGx[2*FH], sBex[2*FH], sGh[2*FH], sBeh[2*FH], sBsum[2*FH];
    __shared__ float sGc[2*Hh], sBec[2*Hh];
    __shared__ float sW1[Hh], sB1[Hh], sWo[Hh];
    __shared__ float sBo;
    __shared__ __align__(32) _Float16 sH[WAVES][16][32];   // h, K-padded to 32 (f16)
    __shared__ float sC[WAVES][16][Hh];                    // cell state
    __shared__ __align__(16) float sG[WAVES][16][FH];      // gate pre-activations

    const int tid = threadIdx.x;

    // ---- block-shared build phase ----
    for (int i = tid; i < 2*FH; i += BS) {
        sGx[i] = g_x[i]; sBex[i] = be_x[i];
        sGh[i] = g_h[i]; sBeh[i] = be_h[i];
        sBsum[i] = b_ih[i] + b_hh[i];
    }
    for (int i = tid; i < 2*Hh; i += BS) { sGc[i] = g_c[i]; sBec[i] = be_c[i]; }
    for (int i = tid; i < Hh;   i += BS) { sW1[i] = W1[i]; sB1[i] = b1[i]; sWo[i] = Wo[i]; }
    if (tid == 0) sBo = bo[0];

    // Build A fragments: one (mat, tile, lane) "lane-row" of 16 halves per iteration.
    // K mapping per lane-half is contiguous: k = k0+e (e<16),
    // with k0 = 0 (lanes 0-15) or 16 (lanes 16-31); k >= 20 is zero padding.
    for (int lr = tid; lr < 3 * NT * 32; lr += BS) {
        int lane = lr & 31;
        int t    = (lr >> 5) % NT;
        int mat  = lr / (NT * 32);
        const float* Ws = (mat == 0) ? Wih : ((mat == 1) ? (Wih + FH*Hh) : (Whh + FH*Hh));
        const float* wrow = Ws + (t * 16 + (lane & 15)) * Hh;
        int k0 = (lane < 16) ? 0 : 16;
        _Float16* dst = &sFrag[lr * 16];
        #pragma unroll
        for (int e = 0; e < 8; ++e)            // k0..k0+7, clamp to zero pad
            dst[e] = (_Float16)((k0 + e < Hh) ? wrow[k0 + e] : 0.0f);
        #pragma unroll
        for (int e = 8; e < 16; ++e) {         // k0+8..k0+15, clamp to zero pad
            int k = k0 + e;
            dst[e] = (_Float16)((k < Hh) ? wrow[k] : 0.0f);
        }
    }
    __syncthreads();

    const int lane = tid & 31;
    const int w    = tid >> 5;
    const int row  = lane & 15;                 // this lane's batch row (N dim of WMMA)
    const int hi8  = (lane >= 16) ? 8 : 0;      // C/D: upper half-lanes hold M = r+8
    const int rowBase = blockIdx.x * ROWS_PER_BLOCK + w * 16;

    // ---- h0 = x*W1 + b1, stored f16 with K-padding (j>=20 -> 0) ----
    {
        float xv = x[rowBase + row];
        if (lane < 16) {
            #pragma unroll
            for (int j = 0; j < 16; ++j)
                sH[w][row][j] = (_Float16)fmaf(xv, sW1[j], sB1[j]);
        } else {
            #pragma unroll
            for (int j = 16; j < Hh; ++j)
                sH[w][row][j] = (_Float16)fmaf(xv, sW1[j], sB1[j]);
            #pragma unroll
            for (int j = Hh; j < 32; ++j)
                sH[w][row][j] = (_Float16)0.0f;
        }
    }

    float dotOut = 0.0f;

    #pragma unroll
    for (int l = 0; l < 2; ++l) {
        // B fragment: h^T, [K=32] x [N=16].  lane<16 holds K=0..15, lane>=16 K=16..31.
        v16h bh = *(const v16h*)&sH[w][row][(lane < 16) ? 0 : 16];

        // ig = Wih[l] @ h : 5 WMMA tiles (K=20 fits one K=32 step)
        v8f acc[NT];
        #pragma unroll
        for (int t = 0; t < NT; ++t) {
            v16h a = *(const v16h*)&sFrag[((l * NT + t) * 32 + lane) * 16];
            v8f  z = {};
            acc[t] = __builtin_amdgcn_wmma_f32_16x16x32_f16(false, a, false, bh,
                                                            (short)0, z, false, false);
        }
        // LayerNorm over 80: 40 local values + partner half-wave via shfl_xor(16)
        float s = 0.0f, q = 0.0f;
        #pragma unroll
        for (int t = 0; t < NT; ++t) {
            #pragma unroll
            for (int r = 0; r < 8; ++r) { float v = acc[t][r]; s += v; q += v * v; }
        }
        s += __shfl_xor(s, 16); q += __shfl_xor(q, 16);
        float mu  = s * (1.0f / FH);
        float inv = rsqrtf(fmaxf(q * (1.0f / FH) - mu * mu, 0.0f) + LN_EPS);

        if (l == 0) {
            // hx == 0 -> LN(Whh@0) == be_h[0]; fold directly, no second matmul.
            #pragma unroll
            for (int t = 0; t < NT; ++t) {
                #pragma unroll
                for (int r = 0; r < 8; ++r) {
                    int m = t * 16 + hi8 + r;
                    acc[t][r] = (acc[t][r] - mu) * inv * sGx[m] + sBex[m]
                              + sBeh[m] + sBsum[m];
                }
            }
        } else {
            // layer 1: hx == h, so reuse bh for the recurrent matmul (mat 2 = Whh[1])
            v8f acc2[NT];
            #pragma unroll
            for (int t = 0; t < NT; ++t) {
                v16h a2 = *(const v16h*)&sFrag[((2 * NT + t) * 32 + lane) * 16];
                v8f  z  = {};
                acc2[t] = __builtin_amdgcn_wmma_f32_16x16x32_f16(false, a2, false, bh,
                                                                 (short)0, z, false, false);
            }
            float s2 = 0.0f, q2 = 0.0f;
            #pragma unroll
            for (int t = 0; t < NT; ++t) {
                #pragma unroll
                for (int r = 0; r < 8; ++r) { float v = acc2[t][r]; s2 += v; q2 += v * v; }
            }
            s2 += __shfl_xor(s2, 16); q2 += __shfl_xor(q2, 16);
            float mu2  = s2 * (1.0f / FH);
            float inv2 = rsqrtf(fmaxf(q2 * (1.0f / FH) - mu2 * mu2, 0.0f) + LN_EPS);
            #pragma unroll
            for (int t = 0; t < NT; ++t) {
                #pragma unroll
                for (int r = 0; r < 8; ++r) {
                    int m  = t * 16 + hi8 + r;
                    int mg = FH + m;
                    acc[t][r] = (acc[t][r]  - mu ) * inv  * sGx[mg] + sBex[mg]
                              + (acc2[t][r] - mu2) * inv2 * sGh[mg] + sBeh[mg]
                              + sBsum[mg];
                }
            }
        }

        // Scatter gate pre-activations to LDS (each lane: 5 tiles x 8 consecutive m)
        #pragma unroll
        for (int t = 0; t < NT; ++t) {
            float4* p = (float4*)&sG[w][row][t * 16 + hi8];
            p[0] = make_float4(acc[t][0], acc[t][1], acc[t][2], acc[t][3]);
            p[1] = make_float4(acc[t][4], acc[t][5], acc[t][6], acc[t][7]);
        }

        // ---- elementwise: regroup i/f/g/o per j, cell update, LN(c), h_new ----
        const int jb = (lane < 16) ? 0 : 10;     // lane-pair splits the 20 j's
        float cvv[10], ovv[10];
        float cs = 0.0f, cq = 0.0f;
        #pragma unroll
        for (int jj = 0; jj < 10; ++jj) {
            int j = jb + jj;
            float iv = sG[w][row][j];
            float fv = sG[w][row][Hh + j];
            float gv = sG[w][row][2 * Hh + j];
            float o_ = sG[w][row][3 * Hh + j];
            float cp = (l == 0) ? 0.0f : sC[w][row][j];
            float c  = sigm(fv + 1.0f) * cp + sigm(iv) * tanh_fast(gv);  // FG_BIAS=1
            if (l == 0) sC[w][row][j] = c;       // only needed as cx for layer 1
            cvv[jj] = c; ovv[jj] = o_;
            cs += c; cq += c * c;
        }
        cs += __shfl_xor(cs, 16); cq += __shfl_xor(cq, 16);
        float cmu  = cs * (1.0f / Hh);
        float cinv = rsqrtf(fmaxf(cq * (1.0f / Hh) - cmu * cmu, 0.0f) + LN_EPS);
        #pragma unroll
        for (int jj = 0; jj < 10; ++jj) {
            int j = jb + jj;
            float hn = sigm(ovv[jj]) *
                       tanh_fast((cvv[jj] - cmu) * cinv * sGc[l * Hh + j] + sBec[l * Hh + j]);
            if (l == 0) sH[w][row][j] = (_Float16)hn;  // B-fragment for layer 1 (pad stays 0)
            if (l == 1) dotOut = fmaf(sWo[j], hn, dotOut);
        }
    }

    // ---- out = Wo . h2 + bo ----
    dotOut += __shfl_xor(dotOut, 16);
    if (lane < 16) out[rowBase + lane] = dotOut + sBo;
}

extern "C" void kernel_launch(void* const* d_in, const int* in_sizes, int n_in,
                              void* d_out, int out_size, void* d_ws, size_t ws_size,
                              hipStream_t stream) {
    const float* x    = (const float*)d_in[0];
    const float* W1   = (const float*)d_in[1];
    const float* b1   = (const float*)d_in[2];
    const float* Wih  = (const float*)d_in[3];
    const float* Whh  = (const float*)d_in[4];
    const float* b_ih = (const float*)d_in[5];
    const float* b_hh = (const float*)d_in[6];
    const float* g_x  = (const float*)d_in[7];
    const float* be_x = (const float*)d_in[8];
    const float* g_h  = (const float*)d_in[9];
    const float* be_h = (const float*)d_in[10];
    const float* g_c  = (const float*)d_in[11];
    const float* be_c = (const float*)d_in[12];
    const float* Wo   = (const float*)d_in[13];
    const float* bo   = (const float*)d_in[14];
    float* out = (float*)d_out;

    int nBlocks = (out_size + ROWS_PER_BLOCK - 1) / ROWS_PER_BLOCK;   // 1e6/64 = 15625 exact
    meta_lstm_kernel<<<nBlocks, BS, 0, stream>>>(x, W1, b1, Wih, Whh, b_ih, b_hh,
                                                 g_x, be_x, g_h, be_h, g_c, be_c,
                                                 Wo, bo, out);
}
